// regression_Transformer_36799279792249
// MI455X (gfx1250) — matrix operand order repaired and checked by reading, not compiled, not run on hardware
//
#include <hip/hip_runtime.h>

#define BB 2
#define SS 1024
#define INN 32
#define DD 1024
#define LL 6
#define HH 16
#define HDD 64
#define FFF 4096

typedef __attribute__((ext_vector_type(16))) __bf16 v16bf;
typedef __attribute__((ext_vector_type(8))) float v8f;

union BFV { v16bf v; unsigned short u[16]; };

__device__ __forceinline__ unsigned short f2bf(float f) {
  unsigned int u = __builtin_bit_cast(unsigned int, f);
  u += 0x7FFFu + ((u >> 16) & 1u);   // round-to-nearest-even
  return (unsigned short)(u >> 16);
}

__device__ __forceinline__ v8f wmma_bf16(v16bf a, v16bf b, v8f c) {
  return __builtin_amdgcn_wmma_f32_16x16x32_bf16(false, a, false, b, (short)0, c,
                                                 false, false);
}

// A-fragment (16x32, MxK): lane holds row r=lane&15.
// lanes 0-15: K = 0..15 ; lanes 16-31: K = 16..31.
// Packed form: per 16x32 tile, [lane][j] contiguous -> one 32B load per lane.
__device__ __forceinline__ v16bf load_a_frag(const float* row, int half, float scale) {
  const float4* p0 = (const float4*)(row + half * 16);
  const float4* p1 = (const float4*)(row + half * 16 + 8);
  float4 f0 = p0[0], f1 = p0[1], f2 = p1[0], f3 = p1[1];
  BFV o;
  o.u[0] = f2bf(f0.x * scale); o.u[1] = f2bf(f0.y * scale);
  o.u[2] = f2bf(f0.z * scale); o.u[3] = f2bf(f0.w * scale);
  o.u[4] = f2bf(f1.x * scale); o.u[5] = f2bf(f1.y * scale);
  o.u[6] = f2bf(f1.z * scale); o.u[7] = f2bf(f1.w * scale);
  o.u[8] = f2bf(f2.x * scale); o.u[9] = f2bf(f2.y * scale);
  o.u[10] = f2bf(f2.z * scale); o.u[11] = f2bf(f2.w * scale);
  o.u[12] = f2bf(f3.x * scale); o.u[13] = f2bf(f3.y * scale);
  o.u[14] = f2bf(f3.z * scale); o.u[15] = f2bf(f3.w * scale);
  return o.v;
}

// A-fragment inverse index map (element (m,k) -> lane half / slot j)
__device__ __forceinline__ void a_slot(int k, int& half, int& j) {
  int kin = k & 31;
  half = (kin >> 4) & 1;
  j = kin & 15;
}

// ---- weight repack: fp32 [L,K,N] row-major -> bf16 B-fragment tiles ----
// tile = (n/16)*(K/32) + (k/32); within tile: lane = ((k>>4)&1)*16 + n%16, pos = k%16
__global__ __launch_bounds__(256) void pack_w_kernel(const float* __restrict__ W,
                                                     unsigned short* __restrict__ out,
                                                     int K, int N, int total) {
  int idx = blockIdx.x * 256 + threadIdx.x;
  if (idx >= total) return;
  int kn = K * N;
  int l = idx / kn;
  int rem = idx - l * kn;
  int k = rem / N;
  int n = rem - k * N;
  int lane = ((k >> 4) & 1) * 16 + (n & 15);
  int pos = k & 15;
  size_t tile = (size_t)(n >> 4) * (K >> 5) + (k >> 5);
  out[(size_t)l * kn + (tile * 32 + lane) * 16 + pos] = f2bf(W[idx]);
}

// ---- activation repack: fp32 [M,K] -> bf16 A-fragment tiles ----
// tile = (m/16)*(K/32) + (k/32); lane = half*16 + m%16, slot j per A layout
__global__ __launch_bounds__(256) void pack_a_kernel(const float* __restrict__ A,
                                                     unsigned short* __restrict__ out,
                                                     int K, int total) {
  int idx = blockIdx.x * 256 + threadIdx.x;
  if (idx >= total) return;
  int m = idx / K;
  int k = idx - m * K;
  int half, j;
  a_slot(k, half, j);
  size_t tile = (size_t)(m >> 4) * (K >> 5) + (k >> 5);
  out[(tile * 32 + half * 16 + (m & 15)) * 16 + j] = f2bf(A[idx]);
}

// ---- K repack for Q*K^T: B-frag [32 d x 16 keys] per (b,h,keytile,dchunk) ----
// out[((b*H+h)*(S/16) + s/16)*1024 + (d/32)*512 + lane*16 + pos]
// lane = ((d>>4)&1)*16 + s%16, pos = d%16
__global__ __launch_bounds__(256) void pack_k_kernel(const float* __restrict__ Km,
                                                     unsigned short* __restrict__ out,
                                                     int total) {
  int idx = blockIdx.x * 256 + threadIdx.x;
  if (idx >= total) return;
  int b = idx / (SS * DD);
  int rem = idx - b * (SS * DD);
  int s = rem / DD;
  int dd = rem - s * DD;
  int hh = dd >> 6;
  int d = dd & 63;
  int lane = ((d >> 4) & 1) * 16 + (s & 15);
  int pos = d & 15;
  size_t base = ((size_t)(b * HH + hh) * (SS >> 4) + (s >> 4)) * 1024 + (d >> 5) * 512;
  out[base + lane * 16 + pos] = f2bf(Km[idx]);
}

// ---- V repack for P*V: B-frag [32 keys x 16 hd] per (b,h,keyblock,hdtile) ----
// out[((b*H+h)*(S/32) + s/32)*2048 + (d/16)*512 + lane*16 + pos]
// lane = ((s>>4)&1)*16 + d%16, pos = s%16
__global__ __launch_bounds__(256) void pack_v_kernel(const float* __restrict__ V,
                                                     unsigned short* __restrict__ out,
                                                     int total) {
  int idx = blockIdx.x * 256 + threadIdx.x;
  if (idx >= total) return;
  int b = idx / (SS * DD);
  int rem = idx - b * (SS * DD);
  int s = rem / DD;
  int dd = rem - s * DD;
  int hh = dd >> 6;
  int d = dd & 63;
  int lane = ((s >> 4) & 1) * 16 + (d & 15);
  int pos = s & 15;
  size_t base = ((size_t)(b * HH + hh) * (SS >> 5) + (s >> 5)) * 2048 + (d >> 4) * 512;
  out[base + lane * 16 + pos] = f2bf(V[idx]);
}

// ---- input projection: h = x @ W_in + b_in + pos ----
__global__ __launch_bounds__(256) void inproj_kernel(const float* __restrict__ x,
                                                     const float* __restrict__ Win,
                                                     const float* __restrict__ bin,
                                                     const float* __restrict__ pos,
                                                     float* __restrict__ h) {
  __shared__ float xs[INN];
  const int tok = blockIdx.y;
  const int d = blockIdx.x * 256 + threadIdx.x;
  if (threadIdx.x < INN) xs[threadIdx.x] = x[(size_t)tok * INN + threadIdx.x];
  __syncthreads();
  const int s = tok & (SS - 1);
  float acc = bin[d] + pos[(size_t)s * DD + d];
#pragma unroll
  for (int i = 0; i < INN; ++i) acc += xs[i] * Win[(size_t)i * DD + d];
  h[(size_t)tok * DD + d] = acc;
}

// ---- bf16 WMMA GEMM: C = act(Ap @ Wp + bias); Ap/Wp packed bf16 fragments ----
// PACKOUT: write result directly as packed A-fragments (for feeding next GEMM)
template <bool RELU, bool PACKOUT>
__global__ __launch_bounds__(256) void gemm_kernel(const unsigned short* __restrict__ Ap,
                                                   const unsigned short* __restrict__ Wp,
                                                   const float* __restrict__ bias,
                                                   float* __restrict__ C,
                                                   unsigned short* __restrict__ Cp,
                                                   int M, int N, int K) {
  const int tid = threadIdx.x;
  const int lane = tid & 31;
  const int w = tid >> 5;
  const int r = lane & 15;
  const int half = lane >> 4;
  const int wn = w & 3;
  const int wm = w >> 2;
  const int mbase = blockIdx.y * 64 + wm * 32;
  const int nbase = blockIdx.x * 128 + wn * 32;
  const int ktiles = K >> 5;
  v8f acc00 = {}, acc01 = {}, acc10 = {}, acc11 = {};
  const unsigned short* at0 = Ap + (size_t)(mbase >> 4) * ktiles * 512 + lane * 16;
  const unsigned short* at1 = Ap + (size_t)((mbase >> 4) + 1) * ktiles * 512 + lane * 16;
  const unsigned short* bt0 = Wp + (size_t)(nbase >> 4) * ktiles * 512 + lane * 16;
  const unsigned short* bt1 = Wp + (size_t)((nbase >> 4) + 1) * ktiles * 512 + lane * 16;
  for (int kt = 0; kt < ktiles; ++kt) {
    v16bf a0 = *(const v16bf*)(at0 + (size_t)kt * 512);
    v16bf a1 = *(const v16bf*)(at1 + (size_t)kt * 512);
    v16bf b0 = *(const v16bf*)(bt0 + (size_t)kt * 512);
    v16bf b1 = *(const v16bf*)(bt1 + (size_t)kt * 512);
    acc00 = wmma_bf16(a0, b0, acc00);
    acc01 = wmma_bf16(a0, b1, acc01);
    acc10 = wmma_bf16(a1, b0, acc10);
    acc11 = wmma_bf16(a1, b1, acc11);
  }
  const int n0 = nbase + r, n1 = nbase + 16 + r;
  const float bv0 = bias[n0], bv1 = bias[n1];
#pragma unroll
  for (int i = 0; i < 8; ++i) {
    int m0 = mbase + half * 8 + i;
    int m1 = m0 + 16;
    float v00 = acc00[i] + bv0, v01 = acc01[i] + bv1;
    float v10 = acc10[i] + bv0, v11 = acc11[i] + bv1;
    if (RELU) {
      v00 = fmaxf(v00, 0.f); v01 = fmaxf(v01, 0.f);
      v10 = fmaxf(v10, 0.f); v11 = fmaxf(v11, 0.f);
    }
    if (PACKOUT) {
      int hs0, js0, hs1, js1;
      a_slot(n0, hs0, js0);
      a_slot(n1, hs1, js1);
      size_t t00 = (size_t)(m0 >> 4) * (N >> 5) + (n0 >> 5);
      size_t t01 = (size_t)(m0 >> 4) * (N >> 5) + (n1 >> 5);
      size_t t10 = (size_t)(m1 >> 4) * (N >> 5) + (n0 >> 5);
      size_t t11 = (size_t)(m1 >> 4) * (N >> 5) + (n1 >> 5);
      Cp[(t00 * 32 + hs0 * 16 + (m0 & 15)) * 16 + js0] = f2bf(v00);
      Cp[(t01 * 32 + hs1 * 16 + (m0 & 15)) * 16 + js1] = f2bf(v01);
      Cp[(t10 * 32 + hs0 * 16 + (m1 & 15)) * 16 + js0] = f2bf(v10);
      Cp[(t11 * 32 + hs1 * 16 + (m1 & 15)) * 16 + js1] = f2bf(v11);
    } else {
      C[(size_t)m0 * N + n0] = v00;
      C[(size_t)m0 * N + n1] = v01;
      C[(size_t)m1 * N + n0] = v10;
      C[(size_t)m1 * N + n1] = v11;
    }
  }
}

// ---- flash attention: one wave per (b, head, 16-query tile) ----
// Q fp32 (converted once per block); K/V pre-packed bf16 B-fragments.
__global__ __launch_bounds__(32) void attn_kernel(const float* __restrict__ Q,
                                                  const unsigned short* __restrict__ Kp,
                                                  const unsigned short* __restrict__ Vp,
                                                  float* __restrict__ O,
                                                  const int* __restrict__ lens) {
  const int lane = threadIdx.x & 31;
  const int r = lane & 15, half = lane >> 4;
  const int qt = blockIdx.x, hh = blockIdx.y, b = blockIdx.z;
  const int len = lens[b];
  const int qbase = qt * 16;
  const float scale = 0.125f;  // 1/sqrt(64), folded into Q
  const float* qrow = Q + ((size_t)(b * SS + qbase + r) * HH + hh) * HDD;
  v16bf alo = load_a_frag(qrow, half, scale);        // d 0..31
  v16bf ahi = load_a_frag(qrow + 32, half, scale);   // d 32..63
  const unsigned short* kpb =
      Kp + (size_t)(b * HH + hh) * (SS >> 4) * 1024 + lane * 16;
  const unsigned short* vpb =
      Vp + (size_t)(b * HH + hh) * (SS >> 5) * 2048 + lane * 16;
  float mrow[8], lrow[8];
  v8f oacc[4] = {{}, {}, {}, {}};
#pragma unroll
  for (int i = 0; i < 8; ++i) { mrow[i] = -1e30f; lrow[i] = 0.f; }
  __shared__ float pls[16 * 32];
  const int nblk = (len + 31) >> 5;
  for (int blk = 0; blk < nblk; ++blk) {
    const int kb = blk * 32;
    v8f s0 = {}, s1 = {};
    {
      const unsigned short* kt = kpb + (size_t)blk * 2048;
      s0 = wmma_bf16(alo, *(const v16bf*)(kt), s0);
      s0 = wmma_bf16(ahi, *(const v16bf*)(kt + 512), s0);
      s1 = wmma_bf16(alo, *(const v16bf*)(kt + 1024), s1);
      s1 = wmma_bf16(ahi, *(const v16bf*)(kt + 1536), s1);
    }
    const bool msk0 = (kb + r) >= len;       // per-lane column mask
    const bool msk1 = (kb + 16 + r) >= len;
#pragma unroll
    for (int i = 0; i < 8; ++i) {
      float x0 = msk0 ? -1e30f : s0[i];
      float x1 = msk1 ? -1e30f : s1[i];
      float vmax = fmaxf(x0, x1);
#pragma unroll
      for (int off = 1; off < 16; off <<= 1)
        vmax = fmaxf(vmax, __shfl_xor(vmax, off, 32));   // stays within 16-lane half
      float newm = fmaxf(mrow[i], vmax);
      float e0 = __expf(x0 - newm);
      float e1 = __expf(x1 - newm);
      float rs = e0 + e1;
#pragma unroll
      for (int off = 1; off < 16; off <<= 1) rs += __shfl_xor(rs, off, 32);
      float alpha = __expf(mrow[i] - newm);
      mrow[i] = newm;
      lrow[i] = lrow[i] * alpha + rs;
      oacc[0][i] *= alpha; oacc[1][i] *= alpha;
      oacc[2][i] *= alpha; oacc[3][i] *= alpha;
      pls[(half * 8 + i) * 32 + r] = e0;        // row m, col = key-in-block
      pls[(half * 8 + i) * 32 + 16 + r] = e1;
    }
    __syncthreads();
    v16bf pa = load_a_frag(&pls[r * 32], half, 1.0f);   // C-layout -> A-layout
    __syncthreads();
    const unsigned short* vt = vpb + (size_t)blk * 2048;
#pragma unroll
    for (int j = 0; j < 4; ++j)
      oacc[j] = wmma_bf16(pa, *(const v16bf*)(vt + j * 512), oacc[j]);
  }
#pragma unroll
  for (int i = 0; i < 8; ++i) {
    float inv = 1.0f / lrow[i];
    int m = qbase + half * 8 + i;
    float* orow = O + ((size_t)(b * SS + m) * HH + hh) * HDD;
#pragma unroll
    for (int j = 0; j < 4; ++j) orow[j * 16 + r] = oacc[j][i] * inv;
  }
}

// ---- fused residual + LayerNorm (in-place on h); res may be null ----
__global__ __launch_bounds__(256) void ln_kernel(float* __restrict__ h,
                                                 const float* __restrict__ res,
                                                 const float* __restrict__ g,
                                                 const float* __restrict__ be) {
  const int row = blockIdx.x;
  const int tid = threadIdx.x;
  float v[4];
  float s = 0.f, s2 = 0.f;
#pragma unroll
  for (int i = 0; i < 4; ++i) {
    int d = tid + i * 256;
    float x = h[(size_t)row * DD + d];
    if (res) x += res[(size_t)row * DD + d];
    v[i] = x; s += x; s2 += x * x;
  }
#pragma unroll
  for (int off = 1; off < 32; off <<= 1) {
    s += __shfl_xor(s, off, 32);
    s2 += __shfl_xor(s2, off, 32);
  }
  __shared__ float aS[8], aS2[8];
  if ((tid & 31) == 0) { aS[tid >> 5] = s; aS2[tid >> 5] = s2; }
  __syncthreads();
  s = 0.f; s2 = 0.f;
#pragma unroll
  for (int i = 0; i < 8; ++i) { s += aS[i]; s2 += aS2[i]; }
  const float mean = s * (1.0f / DD);
  const float var = s2 * (1.0f / DD) - mean * mean;
  const float rstd = rsqrtf(var + 1e-5f);
#pragma unroll
  for (int i = 0; i < 4; ++i) {
    int d = tid + i * 256;
    h[(size_t)row * DD + d] = (v[i] - mean) * rstd * g[d] + be[d];
  }
}

// ---- masked mean pool over valid tokens ----
__global__ __launch_bounds__(256) void pool_kernel(const float* __restrict__ h,
                                                   const int* __restrict__ lens,
                                                   float* __restrict__ pooled) {
  const int b = blockIdx.y;
  const int d = blockIdx.x * 256 + threadIdx.x;
  const int len = lens[b];
  float s = 0.f;
  for (int t = 0; t < len; ++t) s += h[(size_t)(b * SS + t) * DD + d];
  pooled[(size_t)b * DD + d] = s / (float)len;
}

// ---- head: out[b] = pooled[b] . W_out + b_out ----
__global__ __launch_bounds__(256) void head_kernel(const float* __restrict__ pooled,
                                                   const float* __restrict__ Wout,
                                                   const float* __restrict__ bout,
                                                   float* __restrict__ out) {
  const int b = blockIdx.x;
  const int tid = threadIdx.x;
  float s = 0.f;
  for (int d = tid; d < DD; d += 256) s += pooled[(size_t)b * DD + d] * Wout[d];
#pragma unroll
  for (int off = 1; off < 32; off <<= 1) s += __shfl_xor(s, off, 32);
  __shared__ float aS[8];
  if ((tid & 31) == 0) aS[tid >> 5] = s;
  __syncthreads();
  if (tid == 0) {
    float t = 0.f;
    for (int i = 0; i < 8; ++i) t += aS[i];
    out[b] = t + bout[0];
  }
}

extern "C" void kernel_launch(void* const* d_in, const int* in_sizes, int n_in,
                              void* d_out, int out_size, void* d_ws, size_t ws_size,
                              hipStream_t stream) {
  (void)in_sizes; (void)n_in; (void)out_size; (void)ws_size;
  const float* x    = (const float*)d_in[0];
  const int* lens   = (const int*)d_in[1];
  const float* Win  = (const float*)d_in[2];
  const float* bin  = (const float*)d_in[3];
  const float* pos  = (const float*)d_in[4];
  const float* Wq   = (const float*)d_in[5];
  const float* bq   = (const float*)d_in[6];
  const float* Wk   = (const float*)d_in[7];
  const float* bk   = (const float*)d_in[8];
  const float* Wv   = (const float*)d_in[9];
  const float* bv   = (const float*)d_in[10];
  const float* Wo   = (const float*)d_in[11];
  const float* bo   = (const float*)d_in[12];
  const float* W1   = (const float*)d_in[13];
  const float* b1   = (const float*)d_in[14];
  const float* W2   = (const float*)d_in[15];
  const float* b2   = (const float*)d_in[16];
  const float* g1   = (const float*)d_in[17];
  const float* be1  = (const float*)d_in[18];
  const float* g2   = (const float*)d_in[19];
  const float* be2  = (const float*)d_in[20];
  const float* gf   = (const float*)d_in[21];
  const float* bff  = (const float*)d_in[22];
  const float* Wout = (const float*)d_in[23];
  const float* bout = (const float*)d_in[24];

  char* ws = (char*)d_ws;
  size_t off = 0;
  auto alloc = [&](size_t bytes) -> void* {
    void* p = ws + off;
    off += (bytes + 255) & ~(size_t)255;
    return p;
  };
  unsigned short* wq_p = (unsigned short*)alloc((size_t)LL * DD * DD * 2);
  unsigned short* wk_p = (unsigned short*)alloc((size_t)LL * DD * DD * 2);
  unsigned short* wv_p = (unsigned short*)alloc((size_t)LL * DD * DD * 2);
  unsigned short* wo_p = (unsigned short*)alloc((size_t)LL * DD * DD * 2);
  unsigned short* w1_p = (unsigned short*)alloc((size_t)LL * DD * FFF * 2);
  unsigned short* w2_p = (unsigned short*)alloc((size_t)LL * FFF * DD * 2);
  float* h   = (float*)alloc((size_t)BB * SS * DD * 4);
  float* qb  = (float*)alloc((size_t)BB * SS * DD * 4);
  float* kbf = (float*)alloc((size_t)BB * SS * DD * 4);
  float* vbf = (float*)alloc((size_t)BB * SS * DD * 4);
  float* ao  = (float*)alloc((size_t)BB * SS * DD * 4);
  float* tb  = (float*)alloc((size_t)BB * SS * DD * 4);
  float* pooled = (float*)alloc((size_t)BB * DD * 4);
  unsigned short* hp  = (unsigned short*)alloc((size_t)BB * SS * DD * 2);
  unsigned short* aop = (unsigned short*)alloc((size_t)BB * SS * DD * 2);
  unsigned short* f1p = (unsigned short*)alloc((size_t)BB * SS * FFF * 2);
  unsigned short* kp  = (unsigned short*)alloc((size_t)BB * SS * DD * 2);
  unsigned short* vp  = (unsigned short*)alloc((size_t)BB * SS * DD * 2);

  const int M = BB * SS;
  const int totAct = M * DD;              // 2M elements
  const int gAct = (totAct + 255) / 256;

  // ---- pack weights to bf16 fragment tiles (one-shot) ----
  {
    int totD = LL * DD * DD;
    int gD = (totD + 255) / 256;
    pack_w_kernel<<<gD, 256, 0, stream>>>(Wq, wq_p, DD, DD, totD);
    pack_w_kernel<<<gD, 256, 0, stream>>>(Wk, wk_p, DD, DD, totD);
    pack_w_kernel<<<gD, 256, 0, stream>>>(Wv, wv_p, DD, DD, totD);
    pack_w_kernel<<<gD, 256, 0, stream>>>(Wo, wo_p, DD, DD, totD);
    int totF = LL * DD * FFF;
    int gF = (totF + 255) / 256;
    pack_w_kernel<<<gF, 256, 0, stream>>>(W1, w1_p, DD, FFF, totF);
    pack_w_kernel<<<gF, 256, 0, stream>>>(W2, w2_p, FFF, DD, totF);
  }

  inproj_kernel<<<dim3(DD / 256, BB * SS), 256, 0, stream>>>(x, Win, bin, pos, h);

  dim3 gD(DD / 128, M / 64);   // N=1024 GEMMs
  dim3 gF(FFF / 128, M / 64);  // N=4096 GEMM
  for (int l = 0; l < LL; ++l) {
    size_t wofD = (size_t)l * DD * DD;
    size_t wofF = (size_t)l * DD * FFF;
    pack_a_kernel<<<gAct, 256, 0, stream>>>(h, hp, DD, totAct);
    gemm_kernel<false, false><<<gD, 256, 0, stream>>>(hp, wq_p + wofD, bq + (size_t)l * DD, qb, nullptr, M, DD, DD);
    gemm_kernel<false, false><<<gD, 256, 0, stream>>>(hp, wk_p + wofD, bk + (size_t)l * DD, kbf, nullptr, M, DD, DD);
    gemm_kernel<false, false><<<gD, 256, 0, stream>>>(hp, wv_p + wofD, bv + (size_t)l * DD, vbf, nullptr, M, DD, DD);
    pack_k_kernel<<<gAct, 256, 0, stream>>>(kbf, kp, totAct);
    pack_v_kernel<<<gAct, 256, 0, stream>>>(vbf, vp, totAct);
    attn_kernel<<<dim3(SS / 16, HH, BB), 32, 0, stream>>>(qb, kp, vp, ao, lens);
    pack_a_kernel<<<gAct, 256, 0, stream>>>(ao, aop, DD, totAct);
    gemm_kernel<false, false><<<gD, 256, 0, stream>>>(aop, wo_p + wofD, bo + (size_t)l * DD, tb, nullptr, M, DD, DD);
    ln_kernel<<<M, 256, 0, stream>>>(h, tb, g1 + (size_t)l * DD, be1 + (size_t)l * DD);
    pack_a_kernel<<<gAct, 256, 0, stream>>>(h, hp, DD, totAct);
    gemm_kernel<true, true><<<gF, 256, 0, stream>>>(hp, w1_p + wofF, b1 + (size_t)l * FFF, nullptr, f1p, M, FFF, DD);
    gemm_kernel<false, false><<<gD, 256, 0, stream>>>(f1p, w2_p + wofF, b2 + (size_t)l * DD, tb, nullptr, M, DD, FFF);
    ln_kernel<<<M, 256, 0, stream>>>(h, tb, g2 + (size_t)l * DD, be2 + (size_t)l * DD);
  }
  ln_kernel<<<M, 256, 0, stream>>>(h, nullptr, gf, bff);
  pool_kernel<<<dim3(DD / 256, BB), 256, 0, stream>>>(h, lens, pooled);
  head_kernel<<<BB, 256, 0, stream>>>(pooled, Wout, bout, (float*)d_out);
}
